// TemporalHGNN_59545426591934
// MI455X (gfx1250) — matrix operand order repaired and checked by reading, not compiled, not run on hardware
//
#include <hip/hip_runtime.h>
#include <stdint.h>

#define NNODES 32768
#define NEDGES 4096
#define DIM    128
#define LN_EPS 1e-5f
#define KSPLIT 8

// ---- our own vector types (avoid HIP struct-vector quirks in unions) ----
typedef __attribute__((ext_vector_type(16))) __bf16 v16bf;
typedef __attribute__((ext_vector_type(8)))  float  v8f;
typedef __attribute__((ext_vector_type(4)))  unsigned int u32x4;
typedef __attribute__((ext_vector_type(2)))  unsigned int u32x2;
typedef __attribute__((ext_vector_type(4)))  float f32x4;

union FragAB { v16bf v; u32x4 q[2]; };

__device__ __forceinline__ unsigned short f2bf(float f) {
  union { float f; uint32_t u; } x; x.f = f;
  uint32_t u = x.u;
  u += 0x7FFFu + ((u >> 16) & 1u);   // round-to-nearest-even
  return (unsigned short)(u >> 16);
}

// ---------------------------------------------------------------- zero
__global__ void k_zero(float* __restrict__ p, int n) {
  int i = blockIdx.x * 256 + threadIdx.x;
  if (i < n) p[i] = 0.f;
}

// ------------------------------------------------- rowsum of Wp (128x128)
__global__ void k_rowsum_wp(const float* __restrict__ Wp, float* __restrict__ rw) {
  int d = threadIdx.x;
  float s = 0.f;
  for (int j = 0; j < DIM; ++j) s += Wp[d * DIM + j];
  rw[d] = s;
}

// ---- H pass: dv (row sums), de (col sums), Hb bf16, HbT bf16 (transposed)
__global__ __launch_bounds__(256)
void k_prep_H(const float* __restrict__ H, unsigned short* __restrict__ Hb,
              unsigned short* __restrict__ HbT,
              float* __restrict__ dv, float* __restrict__ de)
{
  __shared__ unsigned short tile[64][66];
  __shared__ float dvp[64], dep[64];
  const int n0 = blockIdx.y * 64, e0 = blockIdx.x * 64;
  const int t = threadIdx.x;
  const int rsub = t >> 4;        // 0..15
  const int cq = (t & 15) * 4;    // 0,4,...,60
  if (t < 64) { dvp[t] = 0.f; dep[t] = 0.f; }
  __syncthreads();
  float c0 = 0.f, c1 = 0.f, c2 = 0.f, c3 = 0.f;
#pragma unroll
  for (int i = 0; i < 4; ++i) {
    int r = rsub + i * 16;
    f32x4 h = *(const f32x4*)(H + (size_t)(n0 + r) * NEDGES + e0 + cq);
    unsigned short b0 = f2bf(h[0]), b1 = f2bf(h[1]), b2 = f2bf(h[2]), b3 = f2bf(h[3]);
    u32x2 pk; pk[0] = (uint32_t)b0 | ((uint32_t)b1 << 16);
    pk[1] = (uint32_t)b2 | ((uint32_t)b3 << 16);
    *(u32x2*)(Hb + (size_t)(n0 + r) * NEDGES + e0 + cq) = pk;
    tile[r][cq] = b0; tile[r][cq + 1] = b1; tile[r][cq + 2] = b2; tile[r][cq + 3] = b3;
    atomicAdd(&dvp[r], h[0] + h[1] + h[2] + h[3]);
    c0 += h[0]; c1 += h[1]; c2 += h[2]; c3 += h[3];
  }
  atomicAdd(&dep[cq], c0);     atomicAdd(&dep[cq + 1], c1);
  atomicAdd(&dep[cq + 2], c2); atomicAdd(&dep[cq + 3], c3);
  __syncthreads();
  if (t < 64)       atomicAdd(&dv[n0 + t], dvp[t]);
  else if (t < 128) atomicAdd(&de[e0 + t - 64], dep[t - 64]);
#pragma unroll
  for (int i = 0; i < 4; ++i) {
    int er = rsub + i * 16;
    unsigned short a0 = tile[cq][er], a1 = tile[cq + 1][er];
    unsigned short a2 = tile[cq + 2][er], a3 = tile[cq + 3][er];
    u32x2 pk; pk[0] = (uint32_t)a0 | ((uint32_t)a1 << 16);
    pk[1] = (uint32_t)a2 | ((uint32_t)a3 << 16);
    *(u32x2*)(HbT + (size_t)(e0 + er) * NNODES + n0 + cq) = pk;
  }
}

// ------------------------------------------------- dv^{-1/2}, de^{-1}
__global__ void k_scales(const float* __restrict__ dv, const float* __restrict__ de,
                         float* __restrict__ dvis, float* __restrict__ deinv) {
  int i = blockIdx.x * 256 + threadIdx.x;
  if (i < NNODES) {
    float v = dv[i];
    dvis[i] = v > 0.f ? rsqrtf(fmaxf(v, 1e-12f)) : 0.f;
  } else if (i < NNODES + NEDGES) {
    int e = i - NNODES;
    float v = de[e];
    deinv[e] = v > 0.f ? 1.f / fmaxf(v, 1e-12f) : 0.f;
  }
}

// ------------------------------------------------- f32 -> bf16 copy
__global__ void k_cvt_bf16(const float* __restrict__ s, unsigned short* __restrict__ d, int n) {
  int i = blockIdx.x * 256 + threadIdx.x;
  if (i < n) d[i] = f2bf(s[i]);
}

// --------------------------- x + position encoding -> bf16 [N][128]
__global__ void k_xprep(const float* __restrict__ x, const int* __restrict__ lengths,
                        const float* __restrict__ rw, const float* __restrict__ bp,
                        unsigned short* __restrict__ xb) {
  int i = blockIdx.x * 256 + threadIdx.x;         // over N*128
  int n = i >> 7, d = i & 127;
  int L0 = lengths[0];
  int span = 2 * L0;
  int sample = n / span;
  int Li = lengths[sample];
  int local = n - sample * span;
  int pos = local < Li ? local : local - Li;
  float denom = (float)(Li - 1 > 1 ? Li - 1 : 1);
  float s = sinf(3.14159265358979323846f * (float)pos / denom);
  xb[i] = f2bf(x[i] + s * rw[d] + bp[d]);
}

// =========================== shared GEMM core ===========================
// acc[8] (16x128 strip), A row-major [.][lda], BT row-major [128][ldb].
// Double-buffered fragment pipeline; kend-kbeg must be a multiple of 64.
__device__ __forceinline__ void load_a(FragAB& a, const unsigned short* aptr, int k0) {
  a.q[0] = *(const u32x4*)(aptr + k0);
  a.q[1] = *(const u32x4*)(aptr + k0 + 16);
}
__device__ __forceinline__ void load_b8(FragAB* b, const unsigned short* bbase, int ldb, int k0) {
#pragma unroll
  for (int t = 0; t < 8; ++t) {
    const unsigned short* p = bbase + (size_t)t * 16 * ldb + k0;
    b[t].q[0] = *(const u32x4*)(p);
    b[t].q[1] = *(const u32x4*)(p + 16);
  }
}
__device__ __forceinline__ void wmma8(v8f* acc, const FragAB& a, const FragAB* b) {
#pragma unroll
  for (int t = 0; t < 8; ++t)
    acc[t] = __builtin_amdgcn_wmma_f32_16x16x32_bf16(
        false, a.v, false, b[t].v, (short)0, acc[t], false, false);
}

__device__ __forceinline__ void gemm_core(v8f* acc,
                                          const unsigned short* __restrict__ aptr,
                                          const unsigned short* __restrict__ bbase,
                                          int ldb, int kbeg, int kend)
{
  FragAB a0, a1, b0[8], b1[8];
  load_a(a0, aptr, kbeg);
  load_b8(b0, bbase, ldb, kbeg);
  int k0 = kbeg;
#pragma unroll 1
  for (; k0 + 64 < kend; k0 += 64) {
    __builtin_prefetch(aptr + k0 + 1024, 0, 1);       // A stream (HBM) ahead
    load_a(a1, aptr, k0 + 32);
    load_b8(b1, bbase, ldb, k0 + 32);
    wmma8(acc, a0, b0);
    load_a(a0, aptr, k0 + 64);
    load_b8(b0, bbase, ldb, k0 + 64);
    wmma8(acc, a1, b1);
  }
  load_a(a1, aptr, k0 + 32);
  load_b8(b1, bbase, ldb, k0 + 32);
  wmma8(acc, a0, b0);
  wmma8(acc, a1, b1);
}

// ============ bf16 WMMA GEMM: C[M][128] = A[M][K] * BT[128][K]^T
// out = scaleM[m] * (C + bias[c]); stored TRANSPOSED bf16: outT[c][m]
__global__ __launch_bounds__(256)
void k_gemm_scale_storeT(const unsigned short* __restrict__ A, int lda,
                         const unsigned short* __restrict__ BT, int ldb,
                         const float* __restrict__ bias,      // may be null
                         const float* __restrict__ scaleM,
                         unsigned short* __restrict__ outT, int ldo,
                         int K)
{
  const int wave  = threadIdx.x >> 5;
  const int lane  = threadIdx.x & 31;
  const int lrow  = lane & 15;
  const int khalf = lane >> 4;
  const int m0 = blockIdx.x * 128 + wave * 16;

  v8f acc[8];
#pragma unroll
  for (int t = 0; t < 8; ++t)
#pragma unroll
    for (int j = 0; j < 8; ++j) acc[t][j] = 0.f;

  const unsigned short* aptr  = A + (size_t)(m0 + lrow) * lda + khalf * 8;
  const unsigned short* bbase = BT + (size_t)lrow * ldb + khalf * 8;
  gemm_core(acc, aptr, bbase, ldb, 0, K);

  const int mb = m0 + khalf * 8;           // C layout: lane<16 rows 0..7, else 8..15
  float sm[8];
#pragma unroll
  for (int r = 0; r < 8; ++r) sm[r] = scaleM[mb + r];

#pragma unroll
  for (int t = 0; t < 8; ++t) {
    float bv = bias ? bias[t * 16 + lrow] : 0.f;
#pragma unroll
    for (int r = 0; r < 8; ++r) {
      float v = (acc[t][r] + bv) * sm[r];
      outT[(size_t)(t * 16 + lrow) * ldo + mb + r] = f2bf(v);
    }
  }
}

// ============ split-K bf16 WMMA GEMM: atomic f32 accumulate into Cacc[M][128]
__global__ __launch_bounds__(256)
void k_gemm_splitk_atomic(const unsigned short* __restrict__ A, int lda,
                          const unsigned short* __restrict__ BT, int ldb,
                          float* __restrict__ Cacc, int K)
{
  const int wave  = threadIdx.x >> 5;
  const int lane  = threadIdx.x & 31;
  const int lrow  = lane & 15;
  const int khalf = lane >> 4;
  const int m0 = blockIdx.x * 128 + wave * 16;
  const int kchunk = K / KSPLIT;
  const int kbeg = blockIdx.y * kchunk;

  v8f acc[8];
#pragma unroll
  for (int t = 0; t < 8; ++t)
#pragma unroll
    for (int j = 0; j < 8; ++j) acc[t][j] = 0.f;

  const unsigned short* aptr  = A + (size_t)(m0 + lrow) * lda + khalf * 8;
  const unsigned short* bbase = BT + (size_t)lrow * ldb + khalf * 8;
  gemm_core(acc, aptr, bbase, ldb, kbeg, kbeg + kchunk);

  const int mb = m0 + khalf * 8;
#pragma unroll
  for (int t = 0; t < 8; ++t)
#pragma unroll
    for (int r = 0; r < 8; ++r)
      atomicAdd(&Cacc[(size_t)(mb + r) * DIM + t * 16 + lrow], acc[t][r]);
}

// ------------- finish u: uT[c][e] = bf16(deinv[e] * uacc[e][c])
__global__ void k_finish_u(const float* __restrict__ uacc, const float* __restrict__ deinv,
                           unsigned short* __restrict__ uT) {
  int i = blockIdx.x * 256 + threadIdx.x;     // over E*128
  int e = i >> 7, c = i & 127;
  uT[(size_t)c * NEDGES + e] = f2bf(deinv[e] * uacc[i]);
}

// ============ final GEMM: out = relu(LN(scaleM[m] * (A*BT^T))) , f32 [N][128]
__global__ __launch_bounds__(256)
void k_gemm_ln_relu(const unsigned short* __restrict__ A, int lda,
                    const unsigned short* __restrict__ BT, int ldb,
                    const float* __restrict__ scaleM,
                    const float* __restrict__ gamma, const float* __restrict__ beta,
                    float* __restrict__ out, int K)
{
  const int wave  = threadIdx.x >> 5;
  const int lane  = threadIdx.x & 31;
  const int lrow  = lane & 15;
  const int khalf = lane >> 4;
  const int m0 = blockIdx.x * 128 + wave * 16;

  v8f acc[8];
#pragma unroll
  for (int t = 0; t < 8; ++t)
#pragma unroll
    for (int j = 0; j < 8; ++j) acc[t][j] = 0.f;

  const unsigned short* aptr  = A + (size_t)(m0 + lrow) * lda + khalf * 8;
  const unsigned short* bbase = BT + (size_t)lrow * ldb + khalf * 8;
  gemm_core(acc, aptr, bbase, ldb, 0, K);

  const int mb = m0 + khalf * 8;
  float sm[8];
#pragma unroll
  for (int r = 0; r < 8; ++r) sm[r] = scaleM[mb + r];
  float g[8], bb[8];
#pragma unroll
  for (int t = 0; t < 8; ++t) { g[t] = gamma[t * 16 + lrow]; bb[t] = beta[t * 16 + lrow]; }

  // Row r of this wave's strip lives in acc[0..7][r] (col 16t+lrow), 16 cols per
  // 16-lane half; butterfly across lanes completes the 128-col reduction.
  float mean[8], rstd[8];
#pragma unroll
  for (int r = 0; r < 8; ++r) {
    float s1 = 0.f, s2 = 0.f;
#pragma unroll
    for (int t = 0; t < 8; ++t) {
      float v = acc[t][r] * sm[r];
      acc[t][r] = v;
      s1 += v; s2 += v * v;
    }
#pragma unroll
    for (int m = 1; m < 16; m <<= 1) {
      s1 += __shfl_xor(s1, m, 32);
      s2 += __shfl_xor(s2, m, 32);
    }
    float mu = s1 * (1.f / 128.f);
    mean[r] = mu;
    rstd[r] = rsqrtf(fmaxf(s2 * (1.f / 128.f) - mu * mu, 0.f) + LN_EPS);
  }

#pragma unroll
  for (int t = 0; t < 8; ++t)
#pragma unroll
    for (int r = 0; r < 8; ++r) {
      float v = (acc[t][r] - mean[r]) * rstd[r] * g[t] + bb[t];
      v = fmaxf(v, 0.f);
      out[(size_t)(mb + r) * DIM + t * 16 + lrow] = v;
    }
}

// =====================================================================
extern "C" void kernel_launch(void* const* d_in, const int* in_sizes, int n_in,
                              void* d_out, int out_size, void* d_ws, size_t ws_size,
                              hipStream_t stream) {
  const float* x       = (const float*)d_in[0];
  const float* H       = (const float*)d_in[1];
  const int*   lengths = (const int*)  d_in[2];
  const float* Wp      = (const float*)d_in[3];
  const float* bp      = (const float*)d_in[4];
  const float* Wt      = (const float*)d_in[5];
  const float* bt      = (const float*)d_in[6];
  const float* gamma   = (const float*)d_in[7];
  const float* beta    = (const float*)d_in[8];
  float* out = (float*)d_out;

  char* w = (char*)d_ws;
  auto alloc = [&](size_t bytes) -> char* {
    char* p = w; w += (bytes + 255) & ~(size_t)255; return p;
  };
  unsigned short* Hb   = (unsigned short*)alloc((size_t)NNODES * NEDGES * 2); // 256 MB
  unsigned short* HbT  = (unsigned short*)alloc((size_t)NNODES * NEDGES * 2); // 256 MB
  float* dv    = (float*)alloc((size_t)(NNODES + NEDGES) * 4);
  float* de    = dv + NNODES;
  float* dvis  = (float*)alloc((size_t)(NNODES + NEDGES) * 4);
  float* deinv = dvis + NNODES;
  float* rw    = (float*)alloc(DIM * 4);
  unsigned short* xb  = (unsigned short*)alloc((size_t)NNODES * DIM * 2);
  unsigned short* Wtb = (unsigned short*)alloc((size_t)DIM * DIM * 2);
  unsigned short* zT  = (unsigned short*)alloc((size_t)DIM * NNODES * 2);
  unsigned short* uT  = (unsigned short*)alloc((size_t)DIM * NEDGES * 2);
  float*          uacc= (float*)alloc((size_t)NEDGES * DIM * 4);             // 2 MB
  (void)ws_size; (void)in_sizes; (void)n_in; (void)out_size;

  // 1) zero dv/de and uacc
  k_zero<<<(NNODES + NEDGES + 255) / 256, 256, 0, stream>>>(dv, NNODES + NEDGES);
  k_zero<<<(NEDGES * DIM + 255) / 256, 256, 0, stream>>>(uacc, NEDGES * DIM);
  // 2) rowsum(Wp)
  k_rowsum_wp<<<1, 128, 0, stream>>>(Wp, rw);
  // 3) H pass: dv, de, Hb, HbT
  dim3 gH(NEDGES / 64, NNODES / 64);
  k_prep_H<<<gH, 256, 0, stream>>>(H, Hb, HbT, dv, de);
  // 4) scales
  k_scales<<<(NNODES + NEDGES + 255) / 256, 256, 0, stream>>>(dv, de, dvis, deinv);
  // 5) Wt -> bf16
  k_cvt_bf16<<<(DIM * DIM + 255) / 256, 256, 0, stream>>>(Wt, Wtb, DIM * DIM);
  // 6) x + pe -> bf16
  k_xprep<<<(NNODES * DIM) / 256, 256, 0, stream>>>(x, lengths, rw, bp, xb);
  // 7) zT = (dvis * (xb @ Wtb^T + bt))^T            [128][N]
  k_gemm_scale_storeT<<<NNODES / 128, 256, 0, stream>>>(
      xb, DIM, Wtb, DIM, bt, dvis, zT, NNODES, DIM);
  // 8) uacc += HbT @ z  (split-K over 8 chunks, 256 blocks total)
  dim3 g2(NEDGES / 128, KSPLIT);
  k_gemm_splitk_atomic<<<g2, 256, 0, stream>>>(HbT, NNODES, zT, NNODES, uacc, NNODES);
  //    uT = (deinv * uacc)^T                        [128][E]
  k_finish_u<<<(NEDGES * DIM) / 256, 256, 0, stream>>>(uacc, deinv, uT);
  // 9) out = relu(LN(dvis * (Hb @ u)))              [N][128]
  k_gemm_ln_relu<<<NNODES / 128, 256, 0, stream>>>(
      Hb, NEDGES, uT, NEDGES, dvis, gamma, beta, out, NEDGES);
}
